// DecoderAttn_30382598652305
// MI455X (gfx1250) — compile-verified, matrix operand order checked by reading, not run on hardware
//
#include <hip/hip_runtime.h>
#include <hip/hip_bf16.h>
#include <stdint.h>

typedef __attribute__((ext_vector_type(16))) _Float16 v16h;
typedef __attribute__((ext_vector_type(8)))  _Float16 v8h;
typedef __attribute__((ext_vector_type(8)))  float    v8f;

#define B_    4
#define N1_   2048
#define N2_   2048
#define D_    256
#define KNN_  16
#define HP_   64
#define ROWS_  (B_ * N1_)          /* 8192  */
#define AROWS_ (B_ * N1_ * KNN_)   /* 131072 */
#define EPS_  1e-5f
#define LDP_  40                   /* LDS row pitch in halves: 80B, 16B-aligned */

// ---- CDNA5 WMMA 16-bit operand fragment (16x16x32) -------------------------
// Lane's 16 elements = K runs [kb,kb+8) and [kb+16,kb+24), kb=(lane&16)>>1.
// With K-major LDS rows (80B pitch) each run is one ds_load_b128.
__device__ __forceinline__ v16h frag_load(const _Float16* rowp, int kb) {
  v8h lo = *(const v8h*)(rowp + kb);
  v8h hi = *(const v8h*)(rowp + kb + 16);
  return __builtin_shufflevector(lo, hi, 0, 1, 2, 3, 4, 5, 6, 7,
                                         8, 9, 10, 11, 12, 13, 14, 15);
}

__device__ __forceinline__ void store2(_Float16* dst, float a, float b) {
  union { _Float16 h[2]; uint32_t u; } p;
  p.h[0] = (_Float16)a; p.h[1] = (_Float16)b;
  *(uint32_t*)dst = p.u;     // 4B-aligned by construction -> one ds_store_b32
}

__device__ __forceinline__ v8f wmma_f16(v16h a, v16h b, v8f c) {
  return __builtin_amdgcn_wmma_f32_16x16x32_f16(false, a, false, b, (short)0, c, false, false);
}

// ============================================================================
// Generic GEMM: C[M,N] = A[M,Kd] @ W[Kd,N] + bias (+ optional residual)
// block = 256 thr = 8 waves (2x4) on a 32x128 tile; wave tile 16x32 (2 WMMA/step)
// ============================================================================
__global__ void pt_gemm_bias(const float* __restrict__ A, const float* __restrict__ W,
                             const float* __restrict__ bias, const float* __restrict__ resid,
                             float* __restrict__ C, int M, int Kd, int N) {
  __shared__ _Float16 As[32][LDP_];
  __shared__ _Float16 Wt[128][LDP_];   // transposed: Wt[n][k]
  const int tid  = threadIdx.x;
  const int lane = tid & 31;
  const int wave = tid >> 5;
  const int wm   = wave >> 2;          // 0..1
  const int wn   = wave & 3;           // 0..3
  const int tilesN = N >> 7;
  const int tileM  = (blockIdx.x / tilesN) * 32;
  const int tileN  = (blockIdx.x % tilesN) * 128;
  v8f acc0 = {}, acc1 = {};
  for (int k0 = 0; k0 < Kd; k0 += 32) {
    #pragma unroll
    for (int e = 0; e < 2; ++e) {            // A: 512 half-pairs
      int p = tid + e * 256;
      int r = p >> 4, cp = (p & 15) << 1;
      const float* s = &A[(size_t)(tileM + r) * Kd + k0 + cp];
      store2(&As[r][cp], s[0], s[1]);
    }
    #pragma unroll
    for (int e = 0; e < 8; ++e) {            // W: 2048 half-pairs, transposed
      int p = tid + e * 256;
      int c = p & 127, rp = (p >> 7) << 1;
      store2(&Wt[c][rp], W[(size_t)(k0 + rp) * N + tileN + c],
                         W[(size_t)(k0 + rp + 1) * N + tileN + c]);
    }
    __syncthreads();
    const int kb = (lane & 16) >> 1;
    v16h a  = frag_load(&As[(lane & 15) + wm * 16][0], kb);
    v16h b0 = frag_load(&Wt[(lane & 15) + wn * 32][0], kb);
    v16h b1 = frag_load(&Wt[(lane & 15) + wn * 32 + 16][0], kb);
    acc0 = wmma_f16(a, b0, acc0);
    acc1 = wmma_f16(a, b1, acc1);
    __syncthreads();
  }
  const int col0 = tileN + wn * 32 + (lane & 15);
  const float bv0 = bias[col0], bv1 = bias[col0 + 16];
  const int rbase = tileM + wm * 16 + ((lane >> 4) << 3);
  if (resid) {
    #pragma unroll
    for (int r = 0; r < 8; ++r) {
      size_t ro = (size_t)(rbase + r) * N;
      C[ro + col0]      = acc0[r] + bv0 + resid[ro + col0];
      C[ro + col0 + 16] = acc1[r] + bv1 + resid[ro + col0 + 16];
    }
  } else {
    #pragma unroll
    for (int r = 0; r < 8; ++r) {
      size_t ro = (size_t)(rbase + r) * N;
      C[ro + col0]      = acc0[r] + bv0;
      C[ro + col0 + 16] = acc1[r] + bv1;
    }
  }
}

// ============================================================================
// GEMM with BN(affine)+ReLU on A's K dim, f16 output.
// ss layout: scale[0..Kd-1], shift[Kd..2Kd-1]
// ============================================================================
__global__ void pt_gemm_bnrelu(const float* __restrict__ H, const float* __restrict__ ss,
                               const float* __restrict__ W, const float* __restrict__ bias,
                               _Float16* __restrict__ C16, int M, int Kd, int N) {
  __shared__ _Float16 As[32][LDP_];
  __shared__ _Float16 Wt[128][LDP_];
  const int tid  = threadIdx.x;
  const int lane = tid & 31;
  const int wave = tid >> 5;
  const int wm   = wave >> 2;
  const int wn   = wave & 3;
  const int tilesN = N >> 7;
  const int tileM  = (blockIdx.x / tilesN) * 32;
  const int tileN  = (blockIdx.x % tilesN) * 128;
  v8f acc0 = {}, acc1 = {};
  for (int k0 = 0; k0 < Kd; k0 += 32) {
    #pragma unroll
    for (int e = 0; e < 2; ++e) {
      int p = tid + e * 256;
      int r = p >> 4, cp = (p & 15) << 1;
      int kc = k0 + cp;
      const float* s = &H[(size_t)(tileM + r) * Kd + kc];
      float v0 = fmaxf(s[0] * ss[kc]     + ss[Kd + kc],     0.0f);
      float v1 = fmaxf(s[1] * ss[kc + 1] + ss[Kd + kc + 1], 0.0f);
      store2(&As[r][cp], v0, v1);
    }
    #pragma unroll
    for (int e = 0; e < 8; ++e) {
      int p = tid + e * 256;
      int c = p & 127, rp = (p >> 7) << 1;
      store2(&Wt[c][rp], W[(size_t)(k0 + rp) * N + tileN + c],
                         W[(size_t)(k0 + rp + 1) * N + tileN + c]);
    }
    __syncthreads();
    const int kb = (lane & 16) >> 1;
    v16h a  = frag_load(&As[(lane & 15) + wm * 16][0], kb);
    v16h b0 = frag_load(&Wt[(lane & 15) + wn * 32][0], kb);
    v16h b1 = frag_load(&Wt[(lane & 15) + wn * 32 + 16][0], kb);
    acc0 = wmma_f16(a, b0, acc0);
    acc1 = wmma_f16(a, b1, acc1);
    __syncthreads();
  }
  const int col0 = tileN + wn * 32 + (lane & 15);
  const float bv0 = bias[col0], bv1 = bias[col0 + 16];
  const int rbase = tileM + wm * 16 + ((lane >> 4) << 3);
  #pragma unroll
  for (int r = 0; r < 8; ++r) {
    size_t ro = (size_t)(rbase + r) * N;
    C16[ro + col0]      = (_Float16)(acc0[r] + bv0);
    C16[ro + col0 + 16] = (_Float16)(acc1[r] + bv1);
  }
}

// ============================================================================
// Attn hidden GEMM: A row m composed on the fly: k[nidx]-q+pos; Kd=256, N=64.
// H2[m,64] = A @ Wa1 + ba1.  grid.x = AROWS_/32; wave tile 16x16.
// ============================================================================
__global__ void pt_attn_h(const float* __restrict__ kk, const float* __restrict__ q,
                          const _Float16* __restrict__ pos, const int* __restrict__ nidx,
                          const float* __restrict__ W, const float* __restrict__ bias,
                          float* __restrict__ H2) {
  __shared__ _Float16 As[32][LDP_];
  __shared__ _Float16 Wt[64][LDP_];
  const int tid  = threadIdx.x;
  const int lane = tid & 31;
  const int wave = tid >> 5;
  const int wm   = wave >> 2;
  const int wn   = wave & 3;
  const int tileM = blockIdx.x * 32;
  v8f acc = {};
  for (int k0 = 0; k0 < 256; k0 += 32) {
    #pragma unroll
    for (int e = 0; e < 2; ++e) {            // A: 512 composed half-pairs
      int p = tid + e * 256;
      int r = p >> 4, cp = (p & 15) << 1;
      int m = tileM + r;
      int b = m >> 15;
      int i = (m >> 4) & (N1_ - 1);
      int j = nidx[m];
      int col = k0 + cp;
      size_t kj = (size_t)((b << 11) + j) * D_ + col;
      size_t qi = (size_t)((b << 11) + i) * D_ + col;
      size_t pm = (size_t)m * D_ + col;
      float v0 = kk[kj]     - q[qi]     + (float)pos[pm];
      float v1 = kk[kj + 1] - q[qi + 1] + (float)pos[pm + 1];
      store2(&As[r][cp], v0, v1);
    }
    #pragma unroll
    for (int e = 0; e < 4; ++e) {            // W: 1024 half-pairs, transposed
      int p = tid + e * 256;
      int c = p & 63, rp = (p >> 6) << 1;
      store2(&Wt[c][rp], W[(size_t)(k0 + rp) * HP_ + c],
                         W[(size_t)(k0 + rp + 1) * HP_ + c]);
    }
    __syncthreads();
    const int kb = (lane & 16) >> 1;
    v16h a = frag_load(&As[(lane & 15) + wm * 16][0], kb);
    v16h b = frag_load(&Wt[(lane & 15) + wn * 16][0], kb);
    acc = wmma_f16(a, b, acc);
    __syncthreads();
  }
  const int col = wn * 16 + (lane & 15);
  const float bv = bias[col];
  const int rbase = tileM + wm * 16 + ((lane >> 4) << 3);
  #pragma unroll
  for (int r = 0; r < 8; ++r)
    H2[(size_t)(rbase + r) * HP_ + col] = acc[r] + bv;
}

// ============================================================================
// Farthest point sampling: one 1024-thread block per batch (sequential argmax)
// ============================================================================
__global__ void pt_fps(const float* __restrict__ dec_pc, const float* __restrict__ enc_pc,
                       int* __restrict__ fidx) {
  __shared__ float px[4096], py[4096], pz[4096], mind[4096];
  __shared__ float red[1024];
  __shared__ int   redi[1024];
  __shared__ int   s_last;
  const int b = blockIdx.x, t = threadIdx.x;
  for (int i = t; i < N1_; i += 1024) {
    size_t o = (size_t)(b * N1_ + i) * 3;
    px[i] = dec_pc[o]; py[i] = dec_pc[o + 1]; pz[i] = dec_pc[o + 2];
  }
  for (int i = t; i < N2_; i += 1024) {
    size_t o = (size_t)(b * N2_ + i) * 3;
    px[N1_ + i] = enc_pc[o]; py[N1_ + i] = enc_pc[o + 1]; pz[N1_ + i] = enc_pc[o + 2];
  }
  for (int i = t; i < 4096; i += 1024) mind[i] = 1e10f;
  if (t == 0) { s_last = 0; fidx[b * N1_] = 0; }
  __syncthreads();
  for (int s = 1; s < N1_; ++s) {
    const int last = s_last;
    const float lx = px[last], ly = py[last], lz = pz[last];
    float bm = -1.0f; int bi = 0;
    for (int i = t; i < 4096; i += 1024) {
      float dx = px[i] - lx, dy = py[i] - ly, dz = pz[i] - lz;
      float mv = fminf(mind[i], dx * dx + dy * dy + dz * dz);
      mind[i] = mv;
      if (mv > bm) { bm = mv; bi = i; }
    }
    red[t] = bm; redi[t] = bi;
    __syncthreads();
    for (int off = 512; off > 0; off >>= 1) {
      if (t < off) {
        if (red[t + off] > red[t] || (red[t + off] == red[t] && redi[t + off] < redi[t])) {
          red[t] = red[t + off]; redi[t] = redi[t + off];
        }
      }
      __syncthreads();
    }
    if (t == 0) { s_last = redi[0]; fidx[b * N1_ + s] = redi[0]; }
    __syncthreads();
  }
}

// ============================================================================
// Gather xx_s = concat(dx,ex)[fidx], pc_s = concat(dec_pc,enc_pc)[fidx]
// ============================================================================
__global__ void pt_gather(const float* __restrict__ dx, const float* __restrict__ ex,
                          const float* __restrict__ dec_pc, const float* __restrict__ enc_pc,
                          const int* __restrict__ fidx,
                          float* __restrict__ xx_s, float* __restrict__ pc_s) {
  size_t idx = (size_t)blockIdx.x * 256 + threadIdx.x;
  int c = idx & (D_ - 1);
  int row = (int)(idx >> 8);
  int b = row >> 11;
  int f = fidx[row];
  float v = (f < N1_) ? dx[(size_t)((b << 11) + f) * D_ + c]
                      : ex[(size_t)((b << 11) + (f - N1_)) * D_ + c];
  xx_s[idx] = v;
  if (c < 3) {
    pc_s[(size_t)row * 3 + c] = (f < N1_) ? dec_pc[(size_t)((b << 11) + f) * 3 + c]
                                          : enc_pc[(size_t)((b << 11) + (f - N1_)) * 3 + c];
  }
}

// ============================================================================
// KNN: per-thread top-16 (ascending) insertion sort over 2048 candidates
// ============================================================================
__global__ void pt_knn(const float* __restrict__ dec_pc, const float* __restrict__ pc_s,
                       int* __restrict__ nidx) {
  __shared__ float sx[N1_], sy[N1_], sz[N1_];
  const int b = blockIdx.x >> 3;
  const int i = ((blockIdx.x & 7) << 8) + threadIdx.x;
  for (int j = threadIdx.x; j < N1_; j += 256) {
    size_t o = (size_t)((b << 11) + j) * 3;
    sx[j] = pc_s[o]; sy[j] = pc_s[o + 1]; sz[j] = pc_s[o + 2];
  }
  __syncthreads();
  size_t qo = (size_t)((b << 11) + i) * 3;
  const float qx = dec_pc[qo], qy = dec_pc[qo + 1], qz = dec_pc[qo + 2];
  float bd[KNN_]; int bi[KNN_];
  #pragma unroll
  for (int k = 0; k < KNN_; ++k) { bd[k] = 1e30f; bi[k] = 0; }
  for (int j = 0; j < N1_; ++j) {
    float dx = sx[j] - qx, dy = sy[j] - qy, dz = sz[j] - qz;
    float d = dx * dx + dy * dy + dz * dz;
    if (d < bd[KNN_ - 1]) {
      bd[KNN_ - 1] = d; bi[KNN_ - 1] = j;
      #pragma unroll
      for (int p = KNN_ - 1; p > 0; --p) {
        if (bd[p] < bd[p - 1]) {
          float td = bd[p]; bd[p] = bd[p - 1]; bd[p - 1] = td;
          int   ti = bi[p]; bi[p] = bi[p - 1]; bi[p - 1] = ti;
        }
      }
    }
  }
  #pragma unroll
  for (int k = 0; k < KNN_; ++k) nidx[(size_t)((b << 11) + i) * KNN_ + k] = bi[k];
}

// ============================================================================
// pos hidden: H[m,64] = (pc_s[nidx] - dec_pc) @ Wp1 + bp1   (K=3 dot)
// ============================================================================
__global__ void pt_pos_h(const float* __restrict__ pc_s, const float* __restrict__ dec_pc,
                         const int* __restrict__ nidx, const float* __restrict__ Wp1,
                         const float* __restrict__ bp1, float* __restrict__ H) {
  int m = blockIdx.x * 256 + threadIdx.x;
  if (m >= AROWS_) return;
  int b = m >> 15;
  int i = (m >> 4) & (N1_ - 1);
  int j = nidx[m];
  size_t so = (size_t)((b << 11) + j) * 3;
  size_t qo = (size_t)((b << 11) + i) * 3;
  float x = pc_s[so]     - dec_pc[qo];
  float y = pc_s[so + 1] - dec_pc[qo + 1];
  float z = pc_s[so + 2] - dec_pc[qo + 2];
  #pragma unroll 8
  for (int c = 0; c < HP_; ++c)
    H[(size_t)m * HP_ + c] = x * Wp1[c] + y * Wp1[HP_ + c] + z * Wp1[2 * HP_ + c] + bp1[c];
}

// ============================================================================
// Deterministic per-column sum/sumsq of H[M,64]: one block per column
// ============================================================================
__global__ void pt_colstats(const float* __restrict__ H, int M, float* __restrict__ stats) {
  __shared__ float ssum[256], ssq[256];
  const int c = blockIdx.x;
  float ls = 0.0f, lq = 0.0f;
  for (int r = threadIdx.x; r < M; r += 256) {
    float v = H[(size_t)r * HP_ + c];
    ls += v; lq += v * v;
  }
  ssum[threadIdx.x] = ls; ssq[threadIdx.x] = lq;
  __syncthreads();
  for (int off = 128; off > 0; off >>= 1) {
    if (threadIdx.x < off) {
      ssum[threadIdx.x] += ssum[threadIdx.x + off];
      ssq[threadIdx.x]  += ssq[threadIdx.x + off];
    }
    __syncthreads();
  }
  if (threadIdx.x == 0) { stats[c] = ssum[0]; stats[HP_ + c] = ssq[0]; }
}

__global__ void pt_finstats(const float* __restrict__ stats, const float* __restrict__ g,
                            const float* __restrict__ beta, float* __restrict__ ss, float invM) {
  int c = threadIdx.x;
  if (c >= HP_) return;
  float mean = stats[c] * invM;
  float var  = stats[HP_ + c] * invM - mean * mean;
  float sc   = g[c] * rsqrtf(var + EPS_);
  ss[c]       = sc;
  ss[HP_ + c] = beta[c] - mean * sc;
}

// ============================================================================
// softmax over K neighbors (per channel) + weighted sum of (v+pos)
// ============================================================================
__global__ void pt_combine(const _Float16* __restrict__ attn, const _Float16* __restrict__ pos,
                           const float* __restrict__ vv, const int* __restrict__ nidx,
                           float* __restrict__ x) {
  __shared__ int sj[KNN_];
  const int row = blockIdx.x;
  const int c = threadIdx.x;
  const int b = row >> 11;
  if (c < KNN_) sj[c] = nidx[(size_t)row * KNN_ + c];
  __syncthreads();
  const size_t base = (size_t)row * KNN_ * D_ + c;
  float av[KNN_];
  float mx = -1e30f;
  #pragma unroll
  for (int k = 0; k < KNN_; ++k) { av[k] = (float)attn[base + (size_t)k * D_]; mx = fmaxf(mx, av[k]); }
  float sum = 0.0f;
  #pragma unroll
  for (int k = 0; k < KNN_; ++k) { av[k] = __expf(av[k] - mx); sum += av[k]; }
  const float inv = 1.0f / sum;
  float acc = 0.0f;
  #pragma unroll
  for (int k = 0; k < KNN_; ++k) {
    float vp = vv[(size_t)((b << 11) + sj[k]) * D_ + c] + (float)pos[base + (size_t)k * D_];
    acc += vp * av[k];
  }
  x[(size_t)row * D_ + c] = acc * inv;
}

// ============================================================================
extern "C" void kernel_launch(void* const* d_in, const int* in_sizes, int n_in,
                              void* d_out, int out_size, void* d_ws, size_t ws_size,
                              hipStream_t stream) {
  const float* dec_x   = (const float*)d_in[0];
  const float* dec_pc  = (const float*)d_in[1];
  const float* enc_x   = (const float*)d_in[2];
  const float* enc_pc  = (const float*)d_in[3];
  const float* W_pre1  = (const float*)d_in[4];  const float* b_pre1 = (const float*)d_in[5];
  const float* W_pre2  = (const float*)d_in[6];  const float* b_pre2 = (const float*)d_in[7];
  const float* Wq      = (const float*)d_in[8];  const float* bq     = (const float*)d_in[9];
  const float* Wk      = (const float*)d_in[10]; const float* bk     = (const float*)d_in[11];
  const float* Wv      = (const float*)d_in[12]; const float* bv     = (const float*)d_in[13];
  const float* Wp1     = (const float*)d_in[14]; const float* bp1    = (const float*)d_in[15];
  const float* gp      = (const float*)d_in[16]; const float* betap  = (const float*)d_in[17];
  const float* Wp2     = (const float*)d_in[18]; const float* bp2    = (const float*)d_in[19];
  const float* Wa1     = (const float*)d_in[20]; const float* ba1    = (const float*)d_in[21];
  const float* ga      = (const float*)d_in[22]; const float* betaa  = (const float*)d_in[23];
  const float* Wa2     = (const float*)d_in[24]; const float* ba2    = (const float*)d_in[25];
  const float* W_post1 = (const float*)d_in[26]; const float* b_post1= (const float*)d_in[27];
  const float* W_post2 = (const float*)d_in[28]; const float* b_post2= (const float*)d_in[29];

  uint8_t* wp = (uint8_t*)d_ws;
  auto alloc = [&](size_t bytes) { uint8_t* p = wp; wp += (bytes + 255) & ~(size_t)255; return p; };

  float* f_dx   = (float*)alloc((size_t)ROWS_ * D_ * 4);
  float* f_ex   = (float*)alloc((size_t)ROWS_ * D_ * 4);
  float* f_xxs  = (float*)alloc((size_t)ROWS_ * D_ * 4);
  float* f_q    = (float*)alloc((size_t)ROWS_ * D_ * 4);
  float* f_k    = (float*)alloc((size_t)ROWS_ * D_ * 4);
  float* f_v    = (float*)alloc((size_t)ROWS_ * D_ * 4);
  float* f_x    = (float*)alloc((size_t)ROWS_ * D_ * 4);
  float* f_pcs  = (float*)alloc((size_t)ROWS_ * 3 * 4);
  float* f_h    = (float*)alloc((size_t)AROWS_ * HP_ * 4);
  _Float16* h_pos  = (_Float16*)alloc((size_t)AROWS_ * D_ * 2);
  _Float16* h_attn = (_Float16*)alloc((size_t)AROWS_ * D_ * 2);
  int*   i_fidx = (int*)alloc((size_t)ROWS_ * 4);
  int*   i_nidx = (int*)alloc((size_t)AROWS_ * 4);
  float* f_statsP = (float*)alloc(2 * HP_ * 4);
  float* f_ssP    = (float*)alloc(2 * HP_ * 4);
  float* f_statsA = (float*)alloc(2 * HP_ * 4);
  float* f_ssA    = (float*)alloc(2 * HP_ * 4);

  auto gemm = [&](const float* A, const float* W, const float* bias, const float* resid,
                  float* C, int M, int Kd, int N) {
    pt_gemm_bias<<<dim3((M / 32) * (N / 128)), dim3(256), 0, stream>>>(A, W, bias, resid, C, M, Kd, N);
  };

  // 1) pre-linears
  gemm(dec_x, W_pre1, b_pre1, nullptr, f_dx, ROWS_, D_, D_);
  gemm(enc_x, W_pre2, b_pre2, nullptr, f_ex, ROWS_, D_, D_);
  // 2) farthest point sampling + gather
  pt_fps<<<dim3(B_), dim3(1024), 0, stream>>>(dec_pc, enc_pc, i_fidx);
  pt_gather<<<dim3((ROWS_ * D_) / 256), dim3(256), 0, stream>>>(f_dx, f_ex, dec_pc, enc_pc,
                                                                i_fidx, f_xxs, f_pcs);
  // 3) q, k, v
  gemm(f_dx,  Wq, bq, nullptr, f_q, ROWS_, D_, D_);
  gemm(f_xxs, Wk, bk, nullptr, f_k, ROWS_, D_, D_);
  gemm(f_xxs, Wv, bv, nullptr, f_v, ROWS_, D_, D_);
  // 4) knn
  pt_knn<<<dim3(B_ * (N1_ / 256)), dim3(256), 0, stream>>>(dec_pc, f_pcs, i_nidx);
  // 5) pos MLP: hidden -> BN stats -> BN+ReLU WMMA GEMM -> f16 pos
  pt_pos_h<<<dim3(AROWS_ / 256), dim3(256), 0, stream>>>(f_pcs, dec_pc, i_nidx, Wp1, bp1, f_h);
  pt_colstats<<<dim3(HP_), dim3(256), 0, stream>>>(f_h, AROWS_, f_statsP);
  pt_finstats<<<dim3(1), dim3(64), 0, stream>>>(f_statsP, gp, betap, f_ssP, 1.0f / (float)AROWS_);
  pt_gemm_bnrelu<<<dim3((AROWS_ / 32) * (D_ / 128)), dim3(256), 0, stream>>>(
      f_h, f_ssP, Wp2, bp2, h_pos, AROWS_, HP_, D_);
  // 6) attn MLP: fused-gather hidden -> BN stats -> BN+ReLU WMMA GEMM
  pt_attn_h<<<dim3(AROWS_ / 32), dim3(256), 0, stream>>>(f_k, f_q, h_pos, i_nidx, Wa1, ba1, f_h);
  pt_colstats<<<dim3(HP_), dim3(256), 0, stream>>>(f_h, AROWS_, f_statsA);
  pt_finstats<<<dim3(1), dim3(64), 0, stream>>>(f_statsA, ga, betaa, f_ssA, 1.0f / (float)AROWS_);
  pt_gemm_bnrelu<<<dim3((AROWS_ / 32) * (D_ / 128)), dim3(256), 0, stream>>>(
      f_h, f_ssA, Wa2, ba2, h_attn, AROWS_, HP_, D_);
  // 7) softmax over K + weighted sum -> x
  pt_combine<<<dim3(ROWS_), dim3(256), 0, stream>>>(h_attn, h_pos, f_v, i_nidx, f_x);
  // 8) output heads (tuple: x1, dec_pc, x2, enc_pc)
  float* out = (float*)d_out;
  const size_t o_pc1 = (size_t)ROWS_ * D_;
  const size_t o_x2  = o_pc1 + (size_t)ROWS_ * 3;
  const size_t o_pc2 = o_x2 + (size_t)ROWS_ * D_;
  gemm(f_x, W_post1, b_post1, dec_x, out, ROWS_, D_, D_);
  gemm(f_x, W_post2, b_post2, nullptr, out + o_x2, ROWS_, D_, D_);
  hipMemcpyAsync(out + o_pc1, dec_pc, (size_t)ROWS_ * 3 * 4, hipMemcpyDeviceToDevice, stream);
  hipMemcpyAsync(out + o_pc2, enc_pc, (size_t)B_ * N2_ * 3 * 4, hipMemcpyDeviceToDevice, stream);
  (void)in_sizes; (void)n_in; (void)out_size; (void)ws_size;
}
